// CausalAttention_63136019251228
// MI455X (gfx1250) — compile-verified, matrix operand order checked by reading, not run on hardware
//
#include <hip/hip_runtime.h>
#include <hip/hip_bf16.h>
#include <math.h>

// CDNA5 / gfx1250: wave32, WMMA 16x16x32 bf16 -> f32 accumulate,
// async global->LDS tile fills with double buffering, 128x128 block tiles.

typedef __attribute__((ext_vector_type(16))) __bf16 bf16x16;
typedef __attribute__((ext_vector_type(8)))  __bf16 bf16x8;
typedef __attribute__((ext_vector_type(4)))  __bf16 bf16x4;
typedef __attribute__((ext_vector_type(8)))  float  f32x8;
typedef __attribute__((ext_vector_type(4)))  int    i32x4;

#define LSTR    72      // LDS row stride in bf16 (144B: 16B aligned; 36r mod 64 distinct, r=0..15)
#define KC      64      // K-chunk per LDS fill (2 WMMA K-steps)
#define D_MODEL 1024
#define SEQ     2048
#define BATCH   4

#if defined(__gfx1250__) && __has_builtin(__builtin_amdgcn_global_load_async_to_lds_b128)
#define HAVE_ASYNC_LDS 1
#endif

// 16-byte global -> LDS copy. Async (ASYNCcnt-tracked, no VGPR round trip) on
// gfx1250; plain vector copy fallback otherwise.
__device__ __forceinline__ void copy16_g2l(const void* g, void* l) {
#ifdef HAVE_ASYNC_LDS
  typedef __attribute__((address_space(1))) i32x4* as1_p;
  typedef __attribute__((address_space(3))) i32x4* as3_p;
  __builtin_amdgcn_global_load_async_to_lds_b128(
      (as1_p)(unsigned long long)g,
      (as3_p)(unsigned int)(unsigned long long)l,
      0, 0);
#else
  *(uint4*)l = *(const uint4*)g;
#endif
}

__device__ __forceinline__ void copy_fence() {
#ifdef HAVE_ASYNC_LDS
#if __has_builtin(__builtin_amdgcn_s_wait_asynccnt)
  __builtin_amdgcn_s_wait_asynccnt(0);
#else
  asm volatile("s_wait_asynccnt 0" ::: "memory");
#endif
#endif
}

__device__ __forceinline__ f32x8 wmma_bf16(bf16x16 a, bf16x16 b, f32x8 c) {
  // (neg_a, A, neg_b, B, c_mod, C, reuse_a, reuse_b)
  return __builtin_amdgcn_wmma_f32_16x16x32_bf16(false, a, false, b, (short)0, c, false, false);
}

// Load a 16x32 bf16 fragment (A layout; B tiles are stored [N][K] so the same
// loader works). Two aligned 16B chunks per lane -> two ds_load_b128.
__device__ __forceinline__ bf16x16 load_frag(const __bf16* tile, int lane) {
  const uint4* p = (const uint4*)(tile + (size_t)(lane & 15) * LSTR);
  const int h = lane >> 4;     // lanes 0-15: chunks 0,2; lanes 16-31: chunks 1,3
  union { bf16x16 v; uint4 q[2]; } f;
  f.q[0] = p[h];
  f.q[1] = p[2 + h];
  return f.v;
}

__device__ __forceinline__ void zero_acc(f32x8 acc[4][2]) {
#pragma unroll
  for (int a = 0; a < 4; ++a)
#pragma unroll
    for (int b = 0; b < 2; ++b)
#pragma unroll
      for (int q = 0; q < 8; ++q) acc[a][b][q] = 0.0f;
}

// 2 K-steps x (4x2) WMMA tiles on a filled LDS buffer pair (wave tile 64x32).
__device__ __forceinline__ void mma_tiles(const __bf16* As, const __bf16* Bs,
                                          int wm, int wn, int lane, f32x8 acc[4][2]) {
#pragma unroll
  for (int ks = 0; ks < KC; ks += 32) {
    bf16x16 a[4], b[2];
#pragma unroll
    for (int i = 0; i < 4; ++i)
      a[i] = load_frag(As + (size_t)(wm + 16 * i) * LSTR + ks, lane);
#pragma unroll
    for (int j = 0; j < 2; ++j)
      b[j] = load_frag(Bs + (size_t)(wn + 16 * j) * LSTR + ks, lane);
#pragma unroll
    for (int i = 0; i < 4; ++i)
#pragma unroll
      for (int j = 0; j < 2; ++j)
        acc[i][j] = wmma_bf16(a[i], b[j], acc[i][j]);
  }
}

// Contiguous-row tile fill (128 x KC), one b128 async per 16B chunk.
__device__ __forceinline__ void fill128(const __bf16* srcBase, size_t srcStride,
                                        __bf16* dst, int tid) {
  const int r = tid >> 1, c = (tid & 1) * 32;
  const __bf16* s = srcBase + (size_t)r * srcStride + c;
  __bf16*       d = dst + (size_t)r * LSTR + c;
#pragma unroll
  for (int q = 0; q < 4; ++q) copy16_g2l(s + 8 * q, d + 8 * q);
}

// ---------------------------------------------------------------------------
// Prep 1: x f32 -> bf16 (row-major copy).
// ---------------------------------------------------------------------------
__global__ __launch_bounds__(256) void cvt_x_kernel(const float* __restrict__ X,
                                                    __bf16* __restrict__ Xbf) {
  const size_t idx = ((size_t)blockIdx.x * 256 + threadIdx.x) * 8;
  const float4 a = *(const float4*)(X + idx);
  const float4 b = *(const float4*)(X + idx + 4);
  union { bf16x8 v; } o;
  o.v[0] = (__bf16)a.x; o.v[1] = (__bf16)a.y; o.v[2] = (__bf16)a.z; o.v[3] = (__bf16)a.w;
  o.v[4] = (__bf16)b.x; o.v[5] = (__bf16)b.y; o.v[6] = (__bf16)b.z; o.v[7] = (__bf16)b.w;
  *(bf16x8*)(Xbf + idx) = o.v;
}

// ---------------------------------------------------------------------------
// Prep 2: W[k][n] f32 -> Wt[n][k] bf16 (LDS-tiled transpose), z selects Wq/Wk/Wv.
// ---------------------------------------------------------------------------
__global__ __launch_bounds__(256) void wt_kernel(
    const float* __restrict__ Wq, const float* __restrict__ Wk,
    const float* __restrict__ Wv, __bf16* __restrict__ Wt) {
  __shared__ __bf16 t[32][36];
  const int z = blockIdx.z;
  const float* W = (z == 0) ? Wq : (z == 1) ? Wk : Wv;
  __bf16* Ot = Wt + (size_t)z * D_MODEL * D_MODEL;
  const int k0 = blockIdx.x * 32, n0 = blockIdx.y * 32;
  const int tid = threadIdx.x;
  const int r = tid >> 3, c = (tid & 7) * 4;
  const float4 v = *(const float4*)(W + (size_t)(k0 + r) * D_MODEL + n0 + c);
  t[c + 0][r] = (__bf16)v.x; t[c + 1][r] = (__bf16)v.y;
  t[c + 2][r] = (__bf16)v.z; t[c + 3][r] = (__bf16)v.w;
  __syncthreads();
  bf16x4 o = { t[r][c], t[r][c + 1], t[r][c + 2], t[r][c + 3] };
  *(bf16x4*)(Ot + (size_t)(n0 + r) * D_MODEL + k0 + c) = o;
}

// ---------------------------------------------------------------------------
// Kernel 1: Q/K/V projection (x @ W) from bf16 inputs; all fills are async
// direct copies (A = Xbf rows, B = Wt rows). 128x128 block, double-buffered,
// one barrier/iter. z==2 writes V transposed (Vt[b][n][i]) with b128 stores.
// ---------------------------------------------------------------------------
__global__ __launch_bounds__(256) void qkv_gemm_kernel(
    const __bf16* __restrict__ Xbf, const __bf16* __restrict__ Wt,
    __bf16* __restrict__ Q, __bf16* __restrict__ K, __bf16* __restrict__ Vt) {
  __shared__ __bf16 As[2][128 * LSTR];
  __shared__ __bf16 Bs[2][128 * LSTR];
  const int tid  = threadIdx.x;
  const int lane = tid & 31;
  const int wave = tid >> 5;
  const int wm = (wave & 1) * 64;       // 2 waves in M (64 rows each)
  const int wn = (wave >> 1) * 32;      // 4 waves in N (32 cols each)
  const int m0 = blockIdx.x * 128;
  const int n0 = blockIdx.y * 128;
  const int z  = blockIdx.z;

  const __bf16* Arow = Xbf + (size_t)m0 * D_MODEL;
  const __bf16* Brow = Wt + (size_t)z * D_MODEL * D_MODEL + (size_t)n0 * D_MODEL;

  f32x8 acc[4][2];
  zero_acc(acc);

  int buf = 0;
  fill128(Arow, D_MODEL, As[0], tid);
  fill128(Brow, D_MODEL, Bs[0], tid);
  copy_fence();
  __syncthreads();
  for (int kt = 0; kt < D_MODEL; kt += KC) {
    const int nxt = kt + KC;
    if (nxt < D_MODEL) {
      fill128(Arow + nxt, D_MODEL, As[buf ^ 1], tid);
      fill128(Brow + nxt, D_MODEL, Bs[buf ^ 1], tid);
    }
    mma_tiles(As[buf], Bs[buf], wm, wn, lane, acc);
    copy_fence();
    __syncthreads();
    buf ^= 1;
  }

  const int lm = (lane >> 4) * 8;   // D layout: lanes 0-15 -> M=v, lanes 16-31 -> M=v+8
  const int ln = lane & 15;
  if (z == 2) {
    // V transposed: per (tm,tn), lane's 8 accs are M-consecutive -> one b128.
#pragma unroll
    for (int tm = 0; tm < 4; ++tm)
#pragma unroll
      for (int tn = 0; tn < 2; ++tn) {
        union { bf16x8 v; } pk;
#pragma unroll
        for (int e = 0; e < 8; ++e) pk.v[e] = (__bf16)acc[tm][tn][e];
        const int gm = m0 + wm + tm * 16 + lm;      // 8-aligned
        const int gn = n0 + wn + tn * 16 + ln;
        const int bb = gm >> 11, ii = gm & (SEQ - 1);
        *(bf16x8*)(Vt + ((size_t)bb * D_MODEL + gn) * SEQ + ii) = pk.v;
      }
  } else {
    __bf16* O = (z == 0) ? Q : K;
#pragma unroll
    for (int tm = 0; tm < 4; ++tm)
#pragma unroll
      for (int tn = 0; tn < 2; ++tn)
#pragma unroll
        for (int e = 0; e < 8; ++e) {
          const int gm = m0 + wm + tm * 16 + lm + e;
          const int gn = n0 + wn + tn * 16 + ln;
          O[(size_t)gm * D_MODEL + gn] = (__bf16)acc[tm][tn][e];
        }
  }
}

// ---------------------------------------------------------------------------
// Kernel 2: S = (Q @ K^T) / sqrt(1024), causal mask (-inf above diagonal).
// B = K^T -> [N][K] LDS tile is rows of K. Double-buffered async fills.
// ---------------------------------------------------------------------------
__global__ __launch_bounds__(256) void scores_kernel(
    const __bf16* __restrict__ Q, const __bf16* __restrict__ K,
    float* __restrict__ S) {
  const int i0 = blockIdx.x * 128;
  const int j0 = blockIdx.y * 128;
  if (j0 >= i0 + 128) return;            // fully masked tile
  const int b = blockIdx.z;
  __shared__ __bf16 As[2][128 * LSTR];
  __shared__ __bf16 Bs[2][128 * LSTR];
  const int tid = threadIdx.x, lane = tid & 31, wave = tid >> 5;
  const int wm = (wave & 1) * 64, wn = (wave >> 1) * 32;
  const __bf16* Arow = Q + (size_t)b * SEQ * D_MODEL + (size_t)i0 * D_MODEL;
  const __bf16* Brow = K + (size_t)b * SEQ * D_MODEL + (size_t)j0 * D_MODEL;

  f32x8 acc[4][2];
  zero_acc(acc);

  int buf = 0;
  fill128(Arow, D_MODEL, As[0], tid);
  fill128(Brow, D_MODEL, Bs[0], tid);
  copy_fence();
  __syncthreads();
  for (int kt = 0; kt < D_MODEL; kt += KC) {
    const int nxt = kt + KC;
    if (nxt < D_MODEL) {
      fill128(Arow + nxt, D_MODEL, As[buf ^ 1], tid);
      fill128(Brow + nxt, D_MODEL, Bs[buf ^ 1], tid);
    }
    mma_tiles(As[buf], Bs[buf], wm, wn, lane, acc);
    copy_fence();
    __syncthreads();
    buf ^= 1;
  }

  float* Sb = S + (size_t)b * SEQ * SEQ;
  const int lm = (lane >> 4) * 8;
  const int ln = lane & 15;
  const float scale = 0.03125f;          // 1/sqrt(1024)
#pragma unroll
  for (int tm = 0; tm < 4; ++tm)
#pragma unroll
    for (int tn = 0; tn < 2; ++tn)
#pragma unroll
      for (int e = 0; e < 8; ++e) {
        const int gi = i0 + wm + tm * 16 + lm + e;
        const int gj = j0 + wn + tn * 16 + ln;
        float val = acc[tm][tn][e] * scale;
        if (gj > gi) val = -__builtin_inff();
        Sb[(size_t)gi * SEQ + gj] = val;
      }
}

// ---------------------------------------------------------------------------
// Kernel 3: row softmax over the written region (len = round_up_128(i+1)).
// Reads f32 scores, writes bf16 probabilities (sums kept in f32).
// ---------------------------------------------------------------------------
__global__ __launch_bounds__(256) void softmax_kernel(
    const float* __restrict__ S, __bf16* __restrict__ P) {
  __shared__ float red[256];
  const int row = blockIdx.x;            // 0 .. BATCH*SEQ-1
  const int i   = row & (SEQ - 1);
  const float* p = S + (size_t)row * SEQ;
  __bf16*      q = P + (size_t)row * SEQ;
  const int len = (i & ~127) + 128;      // region written by scores_kernel
  const int tid = threadIdx.x;

  float m = -__builtin_inff();
  for (int j = tid; j < len; j += 256) m = fmaxf(m, p[j]);
  red[tid] = m; __syncthreads();
#pragma unroll
  for (int s = 128; s > 0; s >>= 1) {
    if (tid < s) red[tid] = fmaxf(red[tid], red[tid + s]);
    __syncthreads();
  }
  m = red[0]; __syncthreads();

  float sum = 0.0f;
  for (int j = tid; j < len; j += 256) {
    const float e = __expf(p[j] - m);
    q[j] = (__bf16)e;
    sum += e;
  }
  red[tid] = sum; __syncthreads();
#pragma unroll
  for (int s = 128; s > 0; s >>= 1) {
    if (tid < s) red[tid] += red[tid + s];
    __syncthreads();
  }
  const float inv = 1.0f / red[0];
  for (int j = tid; j < len; j += 256) q[j] = (__bf16)((float)q[j] * inv);
}

// ---------------------------------------------------------------------------
// Kernel 4: O = P @ V using Vt (V transposed): B tile = Vt rows, A tile = P
// rows -> both async direct copies, double buffered. K-loop stops at the
// query tile's diagonal (exact: P is 0 in the masked band).
// ---------------------------------------------------------------------------
__global__ __launch_bounds__(256) void pv_gemm_kernel(
    const __bf16* __restrict__ P, const __bf16* __restrict__ Vt,
    float* __restrict__ O) {
  const int i0 = blockIdx.x * 128;
  const int n0 = blockIdx.y * 128;
  const int b  = blockIdx.z;
  __shared__ __bf16 As[2][128 * LSTR];
  __shared__ __bf16 Bs[2][128 * LSTR];
  const int tid = threadIdx.x, lane = tid & 31, wave = tid >> 5;
  const int wm = (wave & 1) * 64, wn = (wave >> 1) * 32;
  const __bf16* Arow = P + (size_t)b * SEQ * SEQ + (size_t)i0 * SEQ;
  const __bf16* Brow = Vt + (size_t)b * D_MODEL * SEQ + (size_t)n0 * SEQ;

  f32x8 acc[4][2];
  zero_acc(acc);

  const int kend = i0 + 128;             // causal truncation
  int buf = 0;
  fill128(Arow, SEQ, As[0], tid);
  fill128(Brow, SEQ, Bs[0], tid);
  copy_fence();
  __syncthreads();
  for (int kt = 0; kt < kend; kt += KC) {
    const int nxt = kt + KC;
    if (nxt < kend) {
      fill128(Arow + nxt, SEQ, As[buf ^ 1], tid);
      fill128(Brow + nxt, SEQ, Bs[buf ^ 1], tid);
    }
    mma_tiles(As[buf], Bs[buf], wm, wn, lane, acc);
    copy_fence();
    __syncthreads();
    buf ^= 1;
  }

  const int lm = (lane >> 4) * 8;
  const int ln = lane & 15;
#pragma unroll
  for (int tm = 0; tm < 4; ++tm)
#pragma unroll
    for (int tn = 0; tn < 2; ++tn)
#pragma unroll
      for (int e = 0; e < 8; ++e) {
        const int gi = i0 + wm + tm * 16 + lm + e;
        const int gn = n0 + wn + tn * 16 + ln;
        O[((size_t)b * SEQ + gi) * D_MODEL + gn] = acc[tm][tn][e];
      }
}

// ---------------------------------------------------------------------------
// Host launcher. Workspace layout (~166 MB):
//   Xbf bf16 [8192][1024]           16 MB
//   Wt  bf16 [3][1024][1024]         6 MB   (W transposed)
//   Q   bf16 [8192][1024]           16 MB
//   K   bf16 [8192][1024]           16 MB
//   Vt  bf16 [4][1024][2048]        16 MB   (V transposed)
//   S   f32  [4][2048][2048]        64 MB
//   Pbf bf16 [4][2048][2048]        32 MB
// ---------------------------------------------------------------------------
extern "C" void kernel_launch(void* const* d_in, const int* in_sizes, int n_in,
                              void* d_out, int out_size, void* d_ws, size_t ws_size,
                              hipStream_t stream) {
  const float* X  = (const float*)d_in[0];
  const float* Wq = (const float*)d_in[1];
  const float* Wk = (const float*)d_in[2];
  const float* Wv = (const float*)d_in[3];
  float* out = (float*)d_out;

  char* ws = (char*)d_ws;
  const size_t qkv_elems = (size_t)BATCH * SEQ * D_MODEL;     // 8192*1024
  const size_t w_elems   = (size_t)3 * D_MODEL * D_MODEL;
  const size_t s_elems   = (size_t)BATCH * SEQ * SEQ;         // 4*2048*2048
  __bf16* Xbf = (__bf16*)ws;
  __bf16* Wt  = Xbf + qkv_elems;
  __bf16* Q   = Wt + w_elems;
  __bf16* K   = Q + qkv_elems;
  __bf16* Vt  = K + qkv_elems;
  float*  S   = (float*)(Vt + qkv_elems);
  __bf16* Pbf = (__bf16*)(S + s_elems);

  cvt_x_kernel<<<dim3((unsigned)(qkv_elems / (256 * 8))), 256, 0, stream>>>(X, Xbf);
  wt_kernel<<<dim3(D_MODEL / 32, D_MODEL / 32, 3), 256, 0, stream>>>(Wq, Wk, Wv, Wt);
  qkv_gemm_kernel<<<dim3((BATCH * SEQ) / 128, D_MODEL / 128, 3), 256, 0, stream>>>(
      Xbf, Wt, Q, K, Vt);
  scores_kernel<<<dim3(SEQ / 128, SEQ / 128, BATCH), 256, 0, stream>>>(Q, K, S);
  softmax_kernel<<<dim3(BATCH * SEQ), 256, 0, stream>>>(S, Pbf);
  pv_gemm_kernel<<<dim3(SEQ / 128, D_MODEL / 128, BATCH), 256, 0, stream>>>(Pbf, Vt, out);
}
